// TripletLoss_2585570312417
// MI455X (gfx1250) — matrix operand order repaired and checked by reading, not compiled, not run on hardware
//
#include <hip/hip_runtime.h>

// ---------------------------------------------------------------------------
// TripletLoss on MI455X (gfx1250):
//   G = X X^T via bf16 hi/lo-split WMMA (3x v_wmma_f32_16x16x32_bf16 per K-chunk)
//   32-column B panels staged in LDS via async global->LDS copies (ASYNCcnt),
//   double buffered across panels; each wave sweeps two 16x16 C sub-tiles per
//   panel. Deterministic final reduction.
// ---------------------------------------------------------------------------

typedef __attribute__((ext_vector_type(16))) __bf16 v16bf;
typedef __attribute__((ext_vector_type(8)))  float  v8f;

union BF16x16 {
    v16bf v;
    uint4 u[2];
};

static __device__ __forceinline__ unsigned short f32_to_bf16_rne(float f) {
    unsigned int u = __float_as_uint(f);
    unsigned int lsb = (u >> 16) & 1u;
    u += 0x7fffu + lsb;
    return (unsigned short)(u >> 16);
}

static __device__ __forceinline__ float bf16_to_f32(unsigned short h) {
    return __uint_as_float(((unsigned int)h) << 16);
}

#define TL_D 512             // feature dim (fixed by the problem: 8192 x 512)
#define TL_KC (TL_D / 32)    // 16 K-chunks of 32
#define TL_STRIDE (TL_D + 8) // panel row stride in shorts (+16B pad: bank-spread)
#define TL_MARGIN 0.3f
#define TL_EPS 1e-12f

// ---------------------------------------------------------------------------
// Kernel 1: split fp32 -> bf16 (hi) + bf16 (residual lo), row squared norms.
// ---------------------------------------------------------------------------
__global__ __launch_bounds__(256) void tl_prep_kernel(
    const float* __restrict__ X,
    unsigned short* __restrict__ Xhi,
    unsigned short* __restrict__ Xlo,
    float* __restrict__ sq)
{
    const int row = blockIdx.x;
    const size_t base = (size_t)row * TL_D;
    float acc = 0.0f;
    for (int k = threadIdx.x; k < TL_D; k += 256) {
        float x = X[base + k];
        acc += x * x;
        unsigned short h = f32_to_bf16_rne(x);
        float hf = bf16_to_f32(h);
        unsigned short l = f32_to_bf16_rne(x - hf);
        Xhi[base + k] = h;
        Xlo[base + k] = l;
    }
    __shared__ float red[256];
    red[threadIdx.x] = acc;
    __syncthreads();
    for (int s = 128; s > 0; s >>= 1) {
        if (threadIdx.x < s) red[threadIdx.x] += red[threadIdx.x + s];
        __syncthreads();
    }
    if (threadIdx.x == 0) sq[row] = red[0];
}

// ---------------------------------------------------------------------------
// Kernel 2: fused Gram-tile WMMA + distance + hardest pos/neg mining.
// Block = 256 threads = 8 waves; wave w owns rows [i0, i0+16).
// Panel = 32 columns (hi+lo, full K) in LDS, async-staged, double buffered.
// ---------------------------------------------------------------------------
__global__ __launch_bounds__(256, 1) void tl_main_kernel(
    const unsigned short* __restrict__ Xhi,
    const unsigned short* __restrict__ Xlo,
    const float* __restrict__ sq,
    const long long* __restrict__ tgt,
    float* __restrict__ rowloss,
    int n)
{
    // [buf][hi/lo][col][K + pad]  -> 2*2*32*520*2 = 133,120 B
    __shared__ __align__(16) unsigned short smem[2][2][32][TL_STRIDE];

    const int tid     = (int)threadIdx.x;
    const int wave    = tid >> 5;
    const int lane    = tid & 31;
    const int halfsel = lane >> 4;      // 0: lanes 0-15, 1: lanes 16-31
    const int lane16  = lane & 15;
    const int i0      = blockIdx.x * 128 + wave * 16;

    // ---- async stage of one 32-column panel (hi+lo, full K) into LDS buffer
    auto issue_panel = [&](int jt, int buf) {
        const int row = tid >> 3;            // 0..31 panel column
        const int c0  = tid & 7;
        const size_t gro = (size_t)(jt * 32 + row) * TL_D;
#pragma unroll
        for (int i = 0; i < 8; ++i) {
            const int c = c0 + 8 * i;        // 16B chunk id, 0..63
            unsigned lds_hi = (unsigned)(size_t)&smem[buf][0][row][c * 8];
            unsigned lds_lo = (unsigned)(size_t)&smem[buf][1][row][c * 8];
            unsigned long long g_hi = (unsigned long long)(Xhi + gro + c * 8);
            unsigned long long g_lo = (unsigned long long)(Xlo + gro + c * 8);
            asm volatile("global_load_async_to_lds_b128 %0, %1, off"
                         :: "v"(lds_hi), "v"(g_hi) : "memory");
            asm volatile("global_load_async_to_lds_b128 %0, %1, off"
                         :: "v"(lds_lo), "v"(g_lo) : "memory");
        }
    };

    // ---- A fragments: 16x32 bf16, k = 16*halfsel + s (contiguous, same as B), register resident
    const size_t arow_off = (size_t)(i0 + lane16) * TL_D;
    BF16x16 a_hi[TL_KC], a_lo[TL_KC];
#pragma unroll
    for (int kc = 0; kc < TL_KC; ++kc) {
        const int kbase = kc * 32 + halfsel * 16;
        a_hi[kc].u[0] = *(const uint4*)(Xhi + arow_off + kbase);
        a_hi[kc].u[1] = *(const uint4*)(Xhi + arow_off + kbase + 8);
        a_lo[kc].u[0] = *(const uint4*)(Xlo + arow_off + kbase);
        a_lo[kc].u[1] = *(const uint4*)(Xlo + arow_off + kbase + 8);
    }

    // ---- C layout: VGPR r, lanes 0-15 -> (M=r, N=lane); lanes 16-31 -> (M=r+8)
    const int mrow = i0 + halfsel * 8;
    float si[8];
    int   ti[8];
#pragma unroll
    for (int r = 0; r < 8; ++r) {
        si[r] = sq[mrow + r];
        ti[r] = (int)tgt[mrow + r];   // class ids < 512: low 32 bits suffice
    }

    float mp[8], mn[8];
#pragma unroll
    for (int r = 0; r < 8; ++r) { mp[r] = -__builtin_inff(); mn[r] = __builtin_inff(); }

    const int ntiles = n >> 5;               // 32-column panels
    issue_panel(0, 0);

    for (int jt = 0; jt < ntiles; ++jt) {
        const int buf = jt & 1;
        asm volatile("s_wait_asynccnt 0x0" ::: "memory");
        __syncthreads();                     // panel(jt) visible to all waves
        if (jt + 1 < ntiles) issue_panel(jt + 1, buf ^ 1);

#pragma unroll
        for (int s = 0; s < 2; ++s) {        // two 16-column sub-tiles per panel
            const int col = s * 16 + lane16;
            const int j0  = jt * 32 + s * 16;
            const float sqj = sq[j0 + lane16];
            const int   tj  = (int)tgt[j0 + lane16];

            // ---- B fragments from LDS (32x16 bf16)
            v8f c = {};
            BF16x16 bh[2], bl[2];
            {
                const uint4* ph = (const uint4*)&smem[buf][0][col][halfsel * 16];
                const uint4* pl = (const uint4*)&smem[buf][1][col][halfsel * 16];
                bh[0].u[0] = ph[0]; bh[0].u[1] = ph[1];
                bl[0].u[0] = pl[0]; bl[0].u[1] = pl[1];
            }
#pragma unroll
            for (int kc = 0; kc < TL_KC; ++kc) {
                const int cur = kc & 1;
                if (kc + 1 < TL_KC) {
                    const int nk = (kc + 1) * 32 + halfsel * 16;
                    const uint4* ph = (const uint4*)&smem[buf][0][col][nk];
                    const uint4* pl = (const uint4*)&smem[buf][1][col][nk];
                    bh[cur ^ 1].u[0] = ph[0]; bh[cur ^ 1].u[1] = ph[1];
                    bl[cur ^ 1].u[0] = pl[0]; bl[cur ^ 1].u[1] = pl[1];
                }
                // G ~= hi*hi + lo*hi + hi*lo  (drop lo*lo; ~fp32 accuracy)
                c = __builtin_amdgcn_wmma_f32_16x16x32_bf16(false, a_hi[kc].v, false, bh[cur].v,
                                                            (short)0, c, false, false);
                c = __builtin_amdgcn_wmma_f32_16x16x32_bf16(false, a_lo[kc].v, false, bh[cur].v,
                                                            (short)0, c, false, false);
                c = __builtin_amdgcn_wmma_f32_16x16x32_bf16(false, a_hi[kc].v, false, bl[cur].v,
                                                            (short)0, c, false, false);
            }

#pragma unroll
            for (int r = 0; r < 8; ++r) {
                float d2   = si[r] + sqj - 2.0f * c[r];
                float dist = sqrtf(fmaxf(d2, TL_EPS));
                if (ti[r] == tj) mp[r] = fmaxf(mp[r], dist);  // diagonal included (as reference)
                else             mn[r] = fminf(mn[r], dist);
            }
        }
    }

    // ---- reduce across the 16 lanes that share each output row
#pragma unroll
    for (int r = 0; r < 8; ++r) {
        float p = mp[r], q = mn[r];
#pragma unroll
        for (int m = 1; m <= 8; m <<= 1) {
            p = fmaxf(p, __shfl_xor(p, m, 32));
            q = fminf(q, __shfl_xor(q, m, 32));
        }
        if (lane16 == 0) {
            rowloss[mrow + r] = fmaxf(p - q + TL_MARGIN, 0.0f);
        }
    }
}

// ---------------------------------------------------------------------------
// Kernel 3: deterministic mean of per-row losses.
// ---------------------------------------------------------------------------
__global__ __launch_bounds__(256) void tl_finalize_kernel(
    const float* __restrict__ rowloss, float* __restrict__ out, int n)
{
    __shared__ float red[256];
    float acc = 0.0f;
    for (int i = threadIdx.x; i < n; i += 256) acc += rowloss[i];
    red[threadIdx.x] = acc;
    __syncthreads();
    for (int s = 128; s > 0; s >>= 1) {
        if (threadIdx.x < s) red[threadIdx.x] += red[threadIdx.x + s];
        __syncthreads();
    }
    if (threadIdx.x == 0) out[0] = red[0] / (float)n;
}

extern "C" void kernel_launch(void* const* d_in, const int* in_sizes, int n_in,
                              void* d_out, int out_size, void* d_ws, size_t ws_size,
                              hipStream_t stream) {
    (void)n_in; (void)out_size; (void)ws_size;
    const float*     X   = (const float*)d_in[0];       // [n, 512] fp32
    const long long* tgt = (const long long*)d_in[1];   // [n] int64
    const int n = in_sizes[1];                          // 8192

    // workspace layout
    char* ws = (char*)d_ws;
    unsigned short* Xhi = (unsigned short*)ws;                       // n*512*2 B
    unsigned short* Xlo = Xhi + (size_t)n * TL_D;                    // n*512*2 B
    float* sq      = (float*)(Xlo + (size_t)n * TL_D);               // n*4 B
    float* rowloss = sq + n;                                         // n*4 B

    tl_prep_kernel<<<n, 256, 0, stream>>>(X, Xhi, Xlo, sq);
    tl_main_kernel<<<n / 128, 256, 0, stream>>>(Xhi, Xlo, sq, tgt, rowloss, n);
    tl_finalize_kernel<<<1, 256, 0, stream>>>(rowloss, (float*)d_out, n);
}
